// Model_39676907881737
// MI455X (gfx1250) — matrix operand order repaired and checked by reading, not compiled, not run on hardware
//
#include <hip/hip_runtime.h>
#include <math.h>

// Problem constants (from reference)
constexpr int Bsz = 4, H = 8, LQ = 2048, LK = 2048, D = 64, DV = 64;
constexpr float SCALE = 0.125f; // 1/sqrt(64)

constexpr int BLOCK_M = 64;   // queries per workgroup (16 per wave, 4 waves)
constexpr int BLOCK_N = 32;   // keys per inner iteration

// LDS strides in halves; *2 bytes must be 16B aligned, padded vs banks
constexpr int KT_STRIDE = 72; // 64 dims + 8 pad -> 144B rows
constexpr int VT_STRIDE = 40; // 32 keys + 8 pad -> 80B rows

typedef __attribute__((ext_vector_type(16))) _Float16 v16h;
typedef __attribute__((ext_vector_type(8)))  _Float16 v8h;
typedef __attribute__((ext_vector_type(4)))  _Float16 v4h;
typedef __attribute__((ext_vector_type(8)))  float    v8f;
typedef __attribute__((ext_vector_type(4)))  unsigned u32x4;
typedef __attribute__((ext_vector_type(4)))  int      i32x4;
typedef __attribute__((ext_vector_type(8)))  int      i32x8;

union Frag16 { v16h v; v8h h[2]; };

// ---------------------------------------------------------------- helpers ----
__device__ __forceinline__ v16h cvt16(const float* __restrict__ p, float s) {
  v16h r;
#pragma unroll
  for (int i = 0; i < 16; ++i) r[i] = (_Float16)(p[i] * s);
  return r;
}

__device__ __forceinline__ unsigned lds_off(const void* p) {
  // generic LDS pointer: aperture in [63:32], byte offset in [31:0]
  return (unsigned)(unsigned long long)(uintptr_t)p;
}

// TDM 2-D tile load: global (data_size=2B, row stride `stride_el`) -> LDS with
// per-row padding. pad codes per D# group1: interval 2^(c+1) DW, amount c+1 DW.
__device__ __forceinline__ void tdm_load_2d(unsigned ldsaddr, const void* gaddr,
    unsigned tile_d0, unsigned tile_d1, unsigned stride_el,
    unsigned pad_int_c, unsigned pad_amt_c) {
  const unsigned long long ga = (unsigned long long)(uintptr_t)gaddr;
  u32x4 g0;
  g0[0] = 1u;                                   // count=1, user mode
  g0[1] = ldsaddr;                              // lds_addr
  g0[2] = (unsigned)ga;                         // global_addr[31:0]
  g0[3] = (unsigned)(ga >> 32) | (2u << 30);    // global_addr[56:32] | type=2
  i32x8 g1;
  g1[0] = (int)((1u << 16) | (1u << 20) | (pad_int_c << 22) | (pad_amt_c << 25));
  g1[1] = (int)(tile_d0 << 16);                 // tensor_dim0[15:0]   (== tile)
  g1[2] = (int)(tile_d1 << 16);                 // dim0 hi=0 | tensor_dim1[15:0]
  g1[3] = (int)(tile_d0 << 16);                 // dim1 hi=0 | tile_dim0
  g1[4] = (int)(tile_d1);                       // tile_dim1 | tile_dim2=0
  g1[5] = (int)(stride_el);                     // tensor_dim0_stride[31:0]
  g1[6] = 0; g1[7] = 0;
  i32x4 z  = {0, 0, 0, 0};
#if defined(__clang_major__) && (__clang_major__ >= 23)
  i32x8 z8 = {0, 0, 0, 0, 0, 0, 0, 0};
  __builtin_amdgcn_tensor_load_to_lds(g0, g1, z, z, z8, 0);
#else
  __builtin_amdgcn_tensor_load_to_lds(g0, g1, z, z, 0);
#endif
}

// ------------------------------------------------- flash-attention core ----
// One 32-key step: S^T = K·Q^T (4 WMMA), online softmax (register trees + 2
// xor-16 shuffles), P^T B-fragment built via partner exchange, O^T += V^T·P^T.
// All fragments are preloaded into distinct registers so DS latency overlaps
// the matrix pipe and the softmax VALU/TRANS work.
__device__ __forceinline__ void fa_step(
    const _Float16* __restrict__ kl, const _Float16* __restrict__ vt,
    unsigned mw, const v16h* __restrict__ bQ,
    float& mrun, float& lrun, v8f* __restrict__ oC,
    int ln, int hi, int klo) {
  // ---- preload all K fragments (t=key sub-tile, c=dim chunk) ----
  Frag16 aK[4];
#pragma unroll
  for (int t = 0; t < 2; ++t)
#pragma unroll
    for (int c = 0; c < 2; ++c) {
      const _Float16* kp = kl + (t * 16 + ln) * KT_STRIDE + c * 32 + hi * 16;
      aK[t * 2 + c].h[0] = *(const v8h*)(kp);
      aK[t * 2 + c].h[1] = *(const v8h*)(kp + 8);
    }
  // ---- issue all V fragment loads now; latency hides under softmax ----
  Frag16 aV[4];
#pragma unroll
  for (int j = 0; j < 4; ++j) {
    const _Float16* vp = vt + (j * 16 + ln) * VT_STRIDE + hi * 16;
    aV[j].h[0] = *(const v8h*)(vp);
    aV[j].h[1] = *(const v8h*)(vp + 8);
  }

  v8f sC[2];
#pragma unroll
  for (int t = 0; t < 2; ++t) {
    v8f acc = v8f{};
    acc = __builtin_amdgcn_wmma_f32_16x16x32_f16(
        false, aK[t * 2 + 0].v, false, bQ[0], (short)0, acc, false, false);
    acc = __builtin_amdgcn_wmma_f32_16x16x32_f16(
        false, aK[t * 2 + 1].v, false, bQ[1], (short)0, acc, false, false);
    sC[t] = acc;
  }

  float mblk = fmaxf(sC[0][0], sC[1][0]);
#pragma unroll
  for (int r = 1; r < 8; ++r) mblk = fmaxf(mblk, fmaxf(sC[0][r], sC[1][r]));
  mblk = fmaxf(mblk, __shfl_xor(mblk, 16, 32));
  const float mnew  = fmaxf(mrun, mblk);
  const float alpha = __expf(mrun - mnew);

  const unsigned mlo = mw >> (hi * 8);        // bit r -> key 8*hi + r
  const unsigned mhi = mw >> (16 + hi * 8);   // bit r -> key 16 + 8*hi + r
  float p0[8], p1[8], rs = 0.0f;
#pragma unroll
  for (int r = 0; r < 8; ++r) {
    p0[r] = ((mlo >> r) & 1u) ? 0.0f : __expf(sC[0][r] - mnew);
    p1[r] = ((mhi >> r) & 1u) ? 0.0f : __expf(sC[1][r] - mnew);
    rs += p0[r] + p1[r];
  }
  rs += __shfl_xor(rs, 16, 32);
  lrun = lrun * alpha + rs;
  mrun = mnew;

  v16h bP; // lane needs keys 16*hi..16*hi+15; 8 come from the xor-16 partner
#pragma unroll
  for (int r = 0; r < 8; ++r) {
    const float send = hi ? p0[r] : p1[r];
    const float recv = __shfl_xor(send, 16, 32);
    bP[r]     = (_Float16)(hi ? recv : p0[r]);
    bP[8 + r] = (_Float16)(hi ? p1[r] : recv);
  }

#pragma unroll
  for (int j = 0; j < 4; ++j) {
    v8f o = oC[j];
#pragma unroll
    for (int r = 0; r < 8; ++r) o[r] *= alpha;
    oC[j] = __builtin_amdgcn_wmma_f32_16x16x32_f16(
        false, aV[j].v, false, bP, (short)0, o, false, false);
  }
}

// ------------------------------------------------------ prep kernels ----
// f32 -> f16 flat convert (K), 4 elements/thread
__global__ __launch_bounds__(256) void prep_kf16(
    const float* __restrict__ src, _Float16* __restrict__ dst, int n4) {
  int i = blockIdx.x * 256 + threadIdx.x;
  if (i >= n4) return;
  const float4 f = ((const float4*)src)[i];
  v4h h; h[0] = (_Float16)f.x; h[1] = (_Float16)f.y;
  h[2] = (_Float16)f.z; h[3] = (_Float16)f.w;
  ((v4h*)dst)[i] = h;
}

// V [bh][key][dv] f32 -> VT [bh][dv][key] f16, 64x64 LDS-tiled transpose
__global__ __launch_bounds__(256) void prep_vt(
    const float* __restrict__ V, _Float16* __restrict__ VT) {
  __shared__ _Float16 tile[64][72];
  const int bh = blockIdx.y, k0 = blockIdx.x * 64;
  const int r = threadIdx.x >> 2, seg = (threadIdx.x & 3) * 16;
  const float* src = V + ((size_t)bh * LK + k0 + r) * DV + seg;
#pragma unroll
  for (int i = 0; i < 16; ++i) tile[r][seg + i] = (_Float16)src[i];
  __syncthreads();
  _Float16* dst = VT + ((size_t)bh * DV + r) * LK + k0 + seg;
#pragma unroll
  for (int i = 0; i < 16; ++i) dst[i] = tile[seg + i][r];
}

// ------------------------------------------------- main kernel (TDM) ----
__global__ __launch_bounds__(128) void fa_fwd_tdm(
    const float* __restrict__ Q, const _Float16* __restrict__ Kh,
    const _Float16* __restrict__ VTh, const unsigned char* __restrict__ mask,
    float* __restrict__ out) {
  __shared__ __align__(128) _Float16 kl2[2][BLOCK_N * KT_STRIDE];
  __shared__ __align__(128) _Float16 vt2[2][DV * VT_STRIDE];

  const int tid  = threadIdx.x;
  const int wave = tid >> 5, lane = tid & 31;
  const int hi = lane >> 4, ln = lane & 15, klo = hi * 8;
  const int bh = blockIdx.y, b = bh >> 3;
  const int q0 = blockIdx.x * BLOCK_M + wave * 16;

  const _Float16* Kb = Kh  + (size_t)bh * LK * D;
  const _Float16* Vb = VTh + (size_t)bh * DV * LK;
  const unsigned char* Mp = mask + (size_t)b * LK;

  v16h bQ[2];
  {
    const float* qr = Q + ((size_t)bh * LQ + q0 + ln) * D;
#pragma unroll
    for (int c = 0; c < 2; ++c) bQ[c] = cvt16(qr + c * 32 + hi * 16, SCALE);
  }

  float mrun = -INFINITY, lrun = 0.0f;
  v8f oC[4];
#pragma unroll
  for (int j = 0; j < 4; ++j) oC[j] = v8f{};

  // prime buffer 0 (tile kb=0)
  if (wave == 0) {
    tdm_load_2d(lds_off(kl2[0]), Kb, 64, 32, 64, 4, 3);
    tdm_load_2d(lds_off(vt2[0]), Vb, 32, 64, LK, 3, 3);
  }

  for (int kb = 0; kb < LK; kb += 2 * BLOCK_N) {
    // ---- A: prefetch tile kb+32 into buf1, wait buf0, compute buf0 ----
    if (wave == 0) {
      tdm_load_2d(lds_off(kl2[1]), Kb + (size_t)(kb + BLOCK_N) * D, 64, 32, 64, 4, 3);
      tdm_load_2d(lds_off(vt2[1]), Vb + (kb + BLOCK_N), 32, 64, LK, 3, 3);
      __builtin_amdgcn_s_wait_tensorcnt(2);
    }
    __syncthreads();
    {
      const unsigned mw = (unsigned)__ballot(Mp[kb + lane] != 0);
      fa_step(kl2[0], vt2[0], mw, bQ, mrun, lrun, oC, ln, hi, klo);
    }
    __syncthreads();
    // ---- B: prefetch tile kb+64 into buf0 (unless last), wait buf1 ----
    if (wave == 0) {
      if (kb + 2 * BLOCK_N < LK) {
        tdm_load_2d(lds_off(kl2[0]), Kb + (size_t)(kb + 2 * BLOCK_N) * D, 64, 32, 64, 4, 3);
        tdm_load_2d(lds_off(vt2[0]), Vb + (kb + 2 * BLOCK_N), 32, 64, LK, 3, 3);
        __builtin_amdgcn_s_wait_tensorcnt(2);
      } else {
        __builtin_amdgcn_s_wait_tensorcnt(0);
      }
    }
    __syncthreads();
    {
      const unsigned mw = (unsigned)__ballot(Mp[kb + BLOCK_N + lane] != 0);
      fa_step(kl2[1], vt2[1], mw, bQ, mrun, lrun, oC, ln, hi, klo);
    }
    __syncthreads();
  }

  const float inv = 1.0f / lrun;
  float* op = out + ((size_t)bh * LQ + q0 + ln) * DV + hi * 8;
#pragma unroll
  for (int j = 0; j < 4; ++j)
#pragma unroll
    for (int r = 0; r < 8; ++r)
      op[j * 16 + r] = oC[j][r] * inv;
}

// ---------------------------------- fallback (in-loop f32->f16 staging) ----
__global__ __launch_bounds__(128) void fa_fwd_fb(
    const float* __restrict__ Q, const float* __restrict__ K,
    const float* __restrict__ V, const unsigned char* __restrict__ mask,
    float* __restrict__ out) {
  __shared__ __align__(128) _Float16 kl[BLOCK_N * KT_STRIDE];
  __shared__ __align__(128) _Float16 vt[DV * VT_STRIDE];

  const int tid  = threadIdx.x;
  const int wave = tid >> 5, lane = tid & 31;
  const int hi = lane >> 4, ln = lane & 15, klo = hi * 8;
  const int bh = blockIdx.y, b = bh >> 3;
  const int q0 = blockIdx.x * BLOCK_M + wave * 16;

  const float* Kp = K + (size_t)bh * LK * D;
  const float* Vp = V + (size_t)bh * LK * DV;
  const unsigned char* Mp = mask + (size_t)b * LK;

  v16h bQ[2];
  {
    const float* qr = Q + ((size_t)bh * LQ + q0 + ln) * D;
#pragma unroll
    for (int c = 0; c < 2; ++c) bQ[c] = cvt16(qr + c * 32 + hi * 16, SCALE);
  }

  float mrun = -INFINITY, lrun = 0.0f;
  v8f oC[4];
#pragma unroll
  for (int j = 0; j < 4; ++j) oC[j] = v8f{};

  for (int kb = 0; kb < LK; kb += BLOCK_N) {
    {
      const int row = tid >> 2, seg = (tid & 3) * 16;
      const float* src = Kp + (size_t)(kb + row) * D + seg;
      _Float16* dst = kl + row * KT_STRIDE + seg;
#pragma unroll
      for (int i = 0; i < 16; ++i) dst[i] = (_Float16)src[i];
    }
    {
      const int key = tid >> 2, seg = (tid & 3) * 16;
      const float* src = Vp + (size_t)(kb + key) * DV + seg;
#pragma unroll
      for (int i = 0; i < 16; ++i)
        vt[(seg + i) * VT_STRIDE + key] = (_Float16)src[i];
    }
    __syncthreads();
    if (kb + BLOCK_N < LK) {
      if (tid < 32)      __builtin_prefetch(Kp + (size_t)(kb + BLOCK_N + tid) * D, 0, 1);
      else if (tid < 64) __builtin_prefetch(Vp + (size_t)(kb + BLOCK_N + (tid - 32)) * DV, 0, 1);
    }
    const unsigned mw = (unsigned)__ballot(Mp[kb + lane] != 0);
    fa_step(kl, vt, mw, bQ, mrun, lrun, oC, ln, hi, klo);
    __syncthreads();
  }

  const float inv = 1.0f / lrun;
  float* op = out + ((size_t)bh * LQ + q0 + ln) * DV + hi * 8;
#pragma unroll
  for (int j = 0; j < 4; ++j)
#pragma unroll
    for (int r = 0; r < 8; ++r)
      op[j * 16 + r] = oC[j][r] * inv;
}

// ------------------------------------------------------------ launcher ----
extern "C" void kernel_launch(void* const* d_in, const int* in_sizes, int n_in,
                              void* d_out, int out_size, void* d_ws, size_t ws_size,
                              hipStream_t stream) {
  const float* Q = (const float*)d_in[0];
  const float* K = (const float*)d_in[1];
  const float* V = (const float*)d_in[2];
  const unsigned char* M = (const unsigned char*)d_in[3];
  float* out = (float*)d_out;

  constexpr size_t KHBYTES = (size_t)Bsz * H * LK * D * sizeof(_Float16);
  constexpr size_t WS_NEED = KHBYTES * 2; // K(f16) + V^T(f16)
  dim3 grid(LQ / BLOCK_M, Bsz * H);

  if (ws_size >= WS_NEED) {
    _Float16* Kh  = (_Float16*)d_ws;
    _Float16* VTh = (_Float16*)((char*)d_ws + KHBYTES);
    const int n4 = Bsz * H * LK * D / 4;
    prep_kf16<<<(n4 + 255) / 256, 256, 0, stream>>>(K, Kh, n4);
    prep_vt<<<dim3(LK / 64, Bsz * H), 256, 0, stream>>>(V, VTh);
    fa_fwd_tdm<<<grid, 128, 0, stream>>>(Q, Kh, VTh, M, out);
  } else {
    fa_fwd_fb<<<grid, 128, 0, stream>>>(Q, K, V, M, out);
  }
}
